// Transformer_74139725463944
// MI455X (gfx1250) — matrix operand order repaired and checked by reading, not compiled, not run on hardware
//
#include <hip/hip_runtime.h>

#define B_ 2
#define LQ_ 21760
#define C_ 256
#define NHEAD_ 8
#define KPT_ 4      // K sampling points
#define NS_ 4
#define NL_ 6
#define FF_ 1024
#define DK_ 32
#define R_ (B_ * LQ_)   // 43520 rows

typedef __attribute__((ext_vector_type(16))) __bf16 v16bf;
typedef __attribute__((ext_vector_type(8)))  float  v8f;

union FragBF { v16bf v; uint4 q[2]; };

__device__ __forceinline__ unsigned short f2bf(float f) {
  unsigned int u = __float_as_uint(f);
  unsigned int r = u + 0x7FFFu + ((u >> 16) & 1u);   // round-to-nearest-even
  return (unsigned short)(r >> 16);
}

// ---------------------------------------------------------------------------
// bf16 WMMA GEMM:  out[M,N] = A[M,K](bf16) * Bt[N,K](bf16)^T + bias
// One wave -> one 32x32 C tile (2x2 register blocking): per 32-wide K step we
// load 2 A-fragments + 2 B-fragments (8 x b128) and issue 4 WMMAs, so the
// matrix pipe, not the load pipe, is the limiter. fp32 accumulation.
// ---------------------------------------------------------------------------
__global__ __launch_bounds__(256) void gemm_bf16_wmma(
    const unsigned short* __restrict__ A,    // M x K, row-major bf16
    const unsigned short* __restrict__ Bt,   // N x K, bf16 (weight transposed)
    const float* __restrict__ bias,          // N (may be null)
    float* __restrict__ outF,                // M x N fp32 (may be null)
    unsigned short* __restrict__ outB,       // M x N bf16 (may be null)
    int M, int N, int K, int relu)
{
  const int wave = threadIdx.x >> 5;
  const int lane = threadIdx.x & 31;
  const int tile = blockIdx.x * 8 + wave;
  const int ntn  = N >> 5;                   // 32-wide N tiles
  const int ntot = (M >> 5) * ntn;           // 32-tall M tiles
  if (tile >= ntot) return;                  // wave-uniform guard (EXEC all-ones)
  const int tm   = tile / ntn;
  const int tn   = tile - tm * ntn;
  const int half = lane >> 4;
  const int l16  = lane & 15;

  // A lane fragment: K = (e&7) + 8*half + 16*(e>=8)  -> two 16B contiguous loads
  const unsigned short* arow0 = A  + (size_t)(tm * 32 + l16) * K + half * 8;
  const unsigned short* arow1 = arow0 + (size_t)16 * K;
  // B lane fragment: K = (e&7) + 8*half + 16*(e>=8)  -> two 16B contiguous loads
  const unsigned short* brow0 = Bt + (size_t)(tn * 32 + l16) * K + half * 8;
  const unsigned short* brow1 = brow0 + (size_t)16 * K;

  v8f acc00 = {0.f,0.f,0.f,0.f,0.f,0.f,0.f,0.f};
  v8f acc01 = {0.f,0.f,0.f,0.f,0.f,0.f,0.f,0.f};
  v8f acc10 = {0.f,0.f,0.f,0.f,0.f,0.f,0.f,0.f};
  v8f acc11 = {0.f,0.f,0.f,0.f,0.f,0.f,0.f,0.f};

  for (int k0 = 0; k0 < K; k0 += 32) {
    FragBF fa0, fa1, fb0, fb1;
    fa0.q[0] = *(const uint4*)(arow0 + k0);
    fa0.q[1] = *(const uint4*)(arow0 + k0 + 16);
    fa1.q[0] = *(const uint4*)(arow1 + k0);
    fa1.q[1] = *(const uint4*)(arow1 + k0 + 16);
    fb0.q[0] = *(const uint4*)(brow0 + k0);
    fb0.q[1] = *(const uint4*)(brow0 + k0 + 16);
    fb1.q[0] = *(const uint4*)(brow1 + k0);
    fb1.q[1] = *(const uint4*)(brow1 + k0 + 16);
    acc00 = __builtin_amdgcn_wmma_f32_16x16x32_bf16(false, fa0.v, false, fb0.v,
                                                    (short)0, acc00, false, false);
    acc01 = __builtin_amdgcn_wmma_f32_16x16x32_bf16(false, fa0.v, false, fb1.v,
                                                    (short)0, acc01, false, false);
    acc10 = __builtin_amdgcn_wmma_f32_16x16x32_bf16(false, fa1.v, false, fb0.v,
                                                    (short)0, acc10, false, false);
    acc11 = __builtin_amdgcn_wmma_f32_16x16x32_bf16(false, fa1.v, false, fb1.v,
                                                    (short)0, acc11, false, false);
  }

  // C layout per 16x16 sub-tile: VGPR r -> M = base + half*8 + r, N = base + l16
  const int n0 = tn * 32 + l16;
  const int n1 = n0 + 16;
  const int m0 = tm * 32 + half * 8;
  const float bz0 = bias ? bias[n0] : 0.0f;
  const float bz1 = bias ? bias[n1] : 0.0f;
#pragma unroll
  for (int r = 0; r < 8; ++r) {
    float c00 = acc00[r] + bz0;
    float c01 = acc01[r] + bz1;
    float c10 = acc10[r] + bz0;
    float c11 = acc11[r] + bz1;
    if (relu) {
      c00 = fmaxf(c00, 0.0f); c01 = fmaxf(c01, 0.0f);
      c10 = fmaxf(c10, 0.0f); c11 = fmaxf(c11, 0.0f);
    }
    const size_t row0 = (size_t)(m0 + r) * N;
    const size_t row1 = (size_t)(m0 + 16 + r) * N;
    if (outF) {
      outF[row0 + n0] = c00; outF[row0 + n1] = c01;
      outF[row1 + n0] = c10; outF[row1 + n1] = c11;
    }
    if (outB) {
      outB[row0 + n0] = f2bf(c00); outB[row0 + n1] = f2bf(c01);
      outB[row1 + n0] = f2bf(c10); outB[row1 + n1] = f2bf(c11);
    }
  }
}

// ---------------------------------------------------------------------------
// Convert fp32 weights (NL, K, N) -> bf16 transposed (NL, N, K)
// ---------------------------------------------------------------------------
__global__ void wconv_t(const float* __restrict__ src, unsigned short* __restrict__ dst,
                        int K, int N, int layers)
{
  long i   = (long)blockIdx.x * 256 + threadIdx.x;
  long per = (long)K * N;
  if (i >= per * layers) return;
  int  li = (int)(i / per);
  long r  = i - (long)li * per;
  int  k  = (int)(r / N);
  int  n  = (int)(r - (long)k * N);
  dst[(long)li * per + (long)n * K + k] = f2bf(src[i]);
}

// xp = x + pos ; also emit bf16 copy for the WMMA GEMMs
__global__ void addpos_k(const float* __restrict__ x, const float* __restrict__ pos,
                         float* __restrict__ xp, unsigned short* __restrict__ xpb, long n)
{
  long i = (long)blockIdx.x * 256 + threadIdx.x;
  if (i >= n) return;
  float v = x[i] + pos[i];
  xp[i]  = v;
  xpb[i] = f2bf(v);
}

// softmax over 16 contiguous logits per (b,q,head)
__global__ void softmax16_k(float* __restrict__ att, long total)
{
  long i = (long)blockIdx.x * 256 + threadIdx.x;
  if (i >= total) return;
  float* p = att + i * 16;
  float v[16];
  float m = -3.4e38f;
#pragma unroll
  for (int j = 0; j < 16; ++j) { v[j] = p[j]; m = fmaxf(m, v[j]); }
  float s = 0.f;
#pragma unroll
  for (int j = 0; j < 16; ++j) { v[j] = __expf(v[j] - m); s += v[j]; }
  float inv = 1.0f / s;
#pragma unroll
  for (int j = 0; j < 16; ++j) p[j] = v[j] * inv;
}

// ---------------------------------------------------------------------------
// Multi-scale deformable sampling. One wave per (b,q,head); lane = channel.
// v: (B,LQ,C) fp32; off: (B,LQ,NHEAD,NS,K,2); att: (B,LQ,NHEAD,NS*K) softmaxed
// out: bf16 (B,LQ,C) (feeds the Wo WMMA GEMM)
// ---------------------------------------------------------------------------
__global__ __launch_bounds__(256) void deform_sample_k(
    const float* __restrict__ v, const float* __restrict__ off,
    const float* __restrict__ att, const float* __restrict__ ref,
    unsigned short* __restrict__ outb)
{
  const int wave = threadIdx.x >> 5;
  const int lane = threadIdx.x & 31;
  long idx = (long)blockIdx.x * 8 + wave;           // over R_ * NHEAD_
  if (idx >= (long)R_ * NHEAD_) return;
  const int  h  = (int)(idx & (NHEAD_ - 1));
  const long bq = idx >> 3;                          // NHEAD_ == 8
  const float rx = ref[bq * 2 + 0];
  const float ry = ref[bq * 2 + 1];
  const float* offp = off + (bq * NHEAD_ + h) * (NS_ * KPT_ * 2);
  const float* attp = att + (bq * NHEAD_ + h) * (NS_ * KPT_);
  const int  ch   = h * DK_ + lane;
  const long brow = (bq / LQ_) * (long)LQ_;          // b * LQ
  float acc = 0.0f;
  int start = 0;
#pragma unroll
  for (int l = 0; l < NS_; ++l) {
    const int   Wl  = 128 >> l;
    const int   Hl  = 128 >> l;
    const float wl1 = (float)(Wl - 1), hl1 = (float)(Hl - 1);
    const float iwl = 1.0f / (float)Wl, ihl = 1.0f / (float)Hl;
    const float* vb = v + (brow + start) * (long)C_ + ch;
#pragma unroll
    for (int k = 0; k < KPT_; ++k) {
      float ox = offp[(l * KPT_ + k) * 2 + 0];
      float oy = offp[(l * KPT_ + k) * 2 + 1];
      float xs = (rx + ox * iwl) * wl1;
      float ys = (ry + oy * ihl) * hl1;
      float x  = fminf(fmaxf(xs, 0.0f), wl1);
      float y  = fminf(fmaxf(ys, 0.0f), hl1);
      float x0 = floorf(x), y0 = floorf(y);
      float x1 = fminf(x0 + 1.0f, wl1), y1 = fminf(y0 + 1.0f, hl1);
      float wx = x - x0, wy = y - y0;
      int x0i = (int)x0, x1i = (int)x1, y0i = (int)y0, y1i = (int)y1;
      float g00 = vb[((long)y0i * Wl + x0i) * C_];
      float g01 = vb[((long)y0i * Wl + x1i) * C_];
      float g10 = vb[((long)y1i * Wl + x0i) * C_];
      float g11 = vb[((long)y1i * Wl + x1i) * C_];
      float bil = (1.0f - wy) * ((1.0f - wx) * g00 + wx * g01)
                +         wy  * ((1.0f - wx) * g10 + wx * g11);
      acc += attp[l * KPT_ + k] * bil;
    }
    start += Hl * Wl;
  }
  outb[bq * C_ + ch] = f2bf(acc);
}

// LayerNorm over C=256: out = LN(a+b)*gamma + beta ; one block per row, wave32 reduce
__global__ __launch_bounds__(256) void layernorm_k(
    const float* __restrict__ a, const float* __restrict__ b,
    const float* __restrict__ gamma, const float* __restrict__ beta,
    float* __restrict__ outF, unsigned short* __restrict__ outB)
{
  __shared__ float red[8];
  const int    tid  = threadIdx.x;
  const size_t base = (size_t)blockIdx.x * C_;
  float v = a[base + tid] + b[base + tid];

  float s = v;
  for (int o = 16; o > 0; o >>= 1) s += __shfl_xor(s, o, 32);
  if ((tid & 31) == 0) red[tid >> 5] = s;
  __syncthreads();
  float tot = 0.f;
#pragma unroll
  for (int w = 0; w < 8; ++w) tot += red[w];
  const float mean = tot * (1.0f / C_);
  __syncthreads();

  const float d = v - mean;
  s = d * d;
  for (int o = 16; o > 0; o >>= 1) s += __shfl_xor(s, o, 32);
  if ((tid & 31) == 0) red[tid >> 5] = s;
  __syncthreads();
  tot = 0.f;
#pragma unroll
  for (int w = 0; w < 8; ++w) tot += red[w];
  const float rs = rsqrtf(tot * (1.0f / C_) + 1e-5f);

  const float o = d * rs * gamma[tid] + beta[tid];
  if (outF) outF[base + tid] = o;
  if (outB) outB[base + tid] = f2bf(o);
}

// ---------------------------------------------------------------------------
extern "C" void kernel_launch(void* const* d_in, const int* in_sizes, int n_in,
                              void* d_out, int out_size, void* d_ws, size_t ws_size,
                              hipStream_t stream)
{
  const float* src  = (const float*)d_in[0];
  const float* pos  = (const float*)d_in[1];
  const float* ref  = (const float*)d_in[2];
  const float* Woff = (const float*)d_in[3];
  const float* boff = (const float*)d_in[4];
  const float* Wat  = (const float*)d_in[5];
  const float* bat  = (const float*)d_in[6];
  const float* Wv   = (const float*)d_in[7];
  const float* bvp  = (const float*)d_in[8];
  const float* Wo   = (const float*)d_in[9];
  const float* bo   = (const float*)d_in[10];
  const float* W1   = (const float*)d_in[11];
  const float* b1   = (const float*)d_in[12];
  const float* W2   = (const float*)d_in[13];
  const float* b2   = (const float*)d_in[14];
  const float* n1s  = (const float*)d_in[15];
  const float* n1b  = (const float*)d_in[16];
  const float* n2s  = (const float*)d_in[17];
  const float* n2b  = (const float*)d_in[18];
  float* out = (float*)d_out;

  // workspace carving (256B aligned slots)
  char* p = (char*)d_ws;
  auto alloc = [&](size_t bytes) -> char* {
    char* r = p; p += (bytes + 255) & ~(size_t)255; return r;
  };
  const size_t RC4 = (size_t)R_ * C_ * 4;
  const size_t RC2 = (size_t)R_ * C_ * 2;
  float*          X     = (float*)alloc(RC4);                       // layer state
  float*          XP    = (float*)alloc(RC4);                       // x + pos
  unsigned short* XPb   = (unsigned short*)alloc(RC2);
  float*          V     = (float*)alloc(RC4);                       // value proj / attn-out fp32
  float*          OFF   = (float*)alloc(RC4);                       // offsets / ffn-out fp32
  float*          ATT   = (float*)alloc((size_t)R_ * 128 * 4);      // attn logits -> weights
  unsigned short* AOUTb = (unsigned short*)alloc(RC2);              // sampled attn out bf16
  float*          Y     = (float*)alloc(RC4);                       // post-LN1 fp32
  unsigned short* Yb    = (unsigned short*)alloc(RC2);
  unsigned short* Hb    = (unsigned short*)alloc((size_t)R_ * FF_ * 2); // relu hidden bf16
  unsigned short* WvT   = (unsigned short*)alloc((size_t)NL_ * 256 * 256 * 2);
  unsigned short* WoffT = (unsigned short*)alloc((size_t)NL_ * 256 * 256 * 2);
  unsigned short* WatT  = (unsigned short*)alloc((size_t)NL_ * 128 * 256 * 2);
  unsigned short* WoT   = (unsigned short*)alloc((size_t)NL_ * 256 * 256 * 2);
  unsigned short* W1T   = (unsigned short*)alloc((size_t)NL_ * 1024 * 256 * 2);
  unsigned short* W2T   = (unsigned short*)alloc((size_t)NL_ * 256 * 1024 * 2);

  auto cdiv = [](long a, long b) { return (a + b - 1) / b; };

  // weight convert + transpose (all layers), bf16 N-major for contiguous B fragments
  auto wt = [&](const float* s, unsigned short* d, int K, int N) {
    long tot = (long)K * N * NL_;
    wconv_t<<<(unsigned)cdiv(tot, 256), 256, 0, stream>>>(s, d, K, N, NL_);
  };
  wt(Wv,   WvT,   256, 256);
  wt(Woff, WoffT, 256, 256);
  wt(Wat,  WatT,  256, 128);
  wt(Wo,   WoT,   256, 256);
  wt(W1,   W1T,   256, 1024);
  wt(W2,   W2T,   1024, 256);

  auto gemm = [&](const unsigned short* A, const unsigned short* Bt, const float* bias,
                  float* oF, unsigned short* oB, int N, int K, int relu) {
    long tiles = (long)(R_ / 32) * (N / 32);
    gemm_bf16_wmma<<<(unsigned)cdiv(tiles, 8), 256, 0, stream>>>(
        A, Bt, bias, oF, oB, R_, N, K, relu);
  };

  const long n_elems = (long)R_ * C_;
  for (int i = 0; i < NL_; ++i) {
    const float* xin = (i == 0) ? src : X;
    addpos_k<<<(unsigned)cdiv(n_elems, 256), 256, 0, stream>>>(xin, pos, XP, XPb, n_elems);

    gemm(XPb, WvT   + (size_t)i * 256 * 256, bvp  + i * C_,  V,   nullptr, 256, 256, 0);
    gemm(XPb, WoffT + (size_t)i * 256 * 256, boff + i * 256, OFF, nullptr, 256, 256, 0);
    gemm(XPb, WatT  + (size_t)i * 128 * 256, bat  + i * 128, ATT, nullptr, 128, 256, 0);

    softmax16_k<<<(unsigned)cdiv((long)R_ * NHEAD_, 256), 256, 0, stream>>>(
        ATT, (long)R_ * NHEAD_);

    deform_sample_k<<<(unsigned)cdiv((long)R_ * NHEAD_, 8), 256, 0, stream>>>(
        V, OFF, ATT, ref, AOUTb);

    gemm(AOUTb, WoT + (size_t)i * 256 * 256, bo + i * C_, V /*reuse as a32*/, nullptr, 256, 256, 0);

    layernorm_k<<<R_, 256, 0, stream>>>(XP, V, n1s + i * C_, n1b + i * C_, Y, Yb);

    gemm(Yb, W1T + (size_t)i * 1024 * 256, b1 + i * FF_, nullptr, Hb, 1024, 256, 1);
    gemm(Hb, W2T + (size_t)i * 256 * 1024, b2 + i * C_,  OFF /*reuse as ffn out*/, nullptr, 256, 1024, 0);

    float* xo = (i == NL_ - 1) ? out : X;
    layernorm_k<<<R_, 256, 0, stream>>>(Y, OFF, n2s + i * C_, n2b + i * C_, xo, nullptr);
  }
  (void)in_sizes; (void)n_in; (void)out_size; (void)ws_size;
}
